// GRUODEBayes_15281493639237
// MI455X (gfx1250) — matrix operand order repaired and checked by reading, not compiled, not run on hardware
//
#include <hip/hip_runtime.h>
#include <stdint.h>

// ---------------------------------------------------------------- types ----
typedef __attribute__((ext_vector_type(16))) _Float16 v16h;
typedef __attribute__((ext_vector_type(8)))  _Float16 v8h;
typedef __attribute__((ext_vector_type(4)))  _Float16 v4h;
typedef __attribute__((ext_vector_type(8)))  float    v8f;
typedef __attribute__((ext_vector_type(4)))  unsigned int u32x4;
typedef __attribute__((ext_vector_type(8)))  int i32x8;
typedef __attribute__((ext_vector_type(4)))  int i32x4;

#if defined(__HIP_DEVICE_COMPILE__) && __has_builtin(__builtin_amdgcn_tensor_load_to_lds)
#define USE_TDM 1
#else
#define USE_TDM 0
#endif

// ------------------------------------------------------------- constants ---
#define BM 128
#define BN 64
#define BK 32
#define PADH 8              // pad halves per LDS row (8 halves = 4 dwords)
#define LDS_ROW (BK + PADH) // 40 halves = 80B, multiple of 16B

// model dims (compile-time for clean elementwise codegen)
constexpr int cB = 2048, cH = 512, cD = 64, cP = 64, cCO = 32;
constexpr int cLDG = 3 * cH;     // 1536

// ---------------------------------------------------------------- helpers --
__device__ inline float sigmoidf_(float x) { return 1.0f / (1.0f + __expf(-x)); }

__device__ inline void blockReduceAtomicAdd(float v, float* target) {
    __shared__ float red[256];
    red[threadIdx.x] = v;
    __syncthreads();
    for (int s = 128; s > 0; s >>= 1) {
        if ((int)threadIdx.x < s) red[threadIdx.x] += red[threadIdx.x + s];
        __syncthreads();
    }
    if (threadIdx.x == 0) atomicAdd(target, red[0]);
}

// ------------------------------------------------------- WMMA GEMM kernel --
// C[M,N] = act( A[M,K](f32) @ Bt[N,K]^T(f16)  + bias[N]  (+ C if ACC) )
// ACT: 0 none, 1 relu, 2 tanh.  ldc = row stride of C (col-offset fusion).
// REQUIREMENT (all call sites satisfy it): M % BM == 0, K % BK == 0.
// Double-buffered software pipeline: TDM B-tile DMA + batched A loads for
// tile i+1 overlap the WMMA work on tile i.
template<int ACT, bool ACC>
__global__ __launch_bounds__(256)
void wmma_gemm_kernel(const float* __restrict__ A, const _Float16* __restrict__ Bt,
                      const float* __restrict__ bias, float* __restrict__ C,
                      int M, int N, int K, int ldc)
{
    __shared__ alignas(16) _Float16 Ash[2][BM][LDS_ROW];
    __shared__ alignas(16) _Float16 Bsh[2][BN][LDS_ROW];

    const int tid  = threadIdx.x;
    const int lane = tid & 31;
    const int wv   = tid >> 5;          // 0..7 waves
    const int r    = lane & 15;
    const int hi   = lane >> 4;         // 0/1 half-wave

    const int m0 = blockIdx.y * BM;
    const int n0 = blockIdx.x * BN;

    v8f acc[4] = {};

    // Per-thread A-load coordinates: one float4 per sub-tile row band.
    const int arow = tid >> 3;             // 0..31
    const int acol = (tid & 7) << 2;       // 0,4,...,28

    float4 stage[4];

    auto loadA = [&](int k0) {
        #pragma unroll
        for (int it = 0; it < 4; ++it)
            stage[it] = *(const float4*)(A + (size_t)(m0 + it * 32 + arow) * K + k0 + acol);
    };
    auto storeA = [&](int buf) {
        #pragma unroll
        for (int it = 0; it < 4; ++it) {
            v4h h;
            h[0] = (_Float16)stage[it].x; h[1] = (_Float16)stage[it].y;
            h[2] = (_Float16)stage[it].z; h[3] = (_Float16)stage[it].w;
            *(v4h*)&Ash[buf][it * 32 + arow][acol] = h;
        }
    };

#if USE_TDM
    auto tdmB = [&](int k0, int buf) {  // call from wave 0 only
        uint64_t ga = (uint64_t)(uintptr_t)(Bt + (size_t)n0 * K + k0);
        int td0 = K - k0;      // remaining elements along K (OOB -> 0)
        int td1 = N - n0;      // remaining rows (OOB -> 0, handles N=32)
        u32x4 g0;
        g0[0] = 1u;                                      // count=1, user D#
        g0[1] = (unsigned int)(uintptr_t)(&Bsh[buf][0][0]);  // lds_addr
        g0[2] = (unsigned int)(ga & 0xFFFFFFFFu);        // global_addr lo
        g0[3] = ((unsigned int)(ga >> 32) & 0x01FFFFFFu) // global_addr hi
                | (2u << 30);                            // type = image
        i32x8 g1;
        g1[0] = (1 << 16)    // data_size = 2 bytes
              | (1 << 20)    // pad_enable
              | (3 << 22)    // pad_interval: 16 dwords (= one 32-half row)
              | (3 << 25);   // pad_amount: 4 dwords (= 8 halves)
        g1[1] = (td0 & 0xFFFF) << 16;
        g1[2] = ((td0 >> 16) & 0xFFFF) | ((td1 & 0xFFFF) << 16);
        g1[3] = ((td1 >> 16) & 0xFFFF) | (BK << 16);     // tile_dim0 = 32
        g1[4] = BN;                                      // tile_dim1 = 64
        g1[5] = K;                                       // dim0 stride lo
        g1[6] = 0;
        g1[7] = 0;
        i32x4 z4 = {};
#if __clang_major__ >= 23
        i32x8 z8 = {};
        __builtin_amdgcn_tensor_load_to_lds(g0, g1, z4, z4, z8, 0);
#else
        __builtin_amdgcn_tensor_load_to_lds(g0, g1, z4, z4, 0);
#endif
    };
#else
    auto loadB = [&](int k0, int buf) {  // fallback: all threads, N-guarded
        #pragma unroll
        for (int it = 0; it < (BN * BK) / 2 / 256; ++it) {
            int idx = it * 256 + tid;
            int row = idx >> 4;
            int kp  = (idx & 15) << 1;
            _Float16 b0 = (_Float16)0.0f, b1 = (_Float16)0.0f;
            int gn = n0 + row;
            if (gn < N) {
                const _Float16* bp = Bt + (size_t)gn * K + k0 + kp;
                b0 = bp[0]; b1 = bp[1];
            }
            Bsh[buf][row][kp]     = b0;
            Bsh[buf][row][kp + 1] = b1;
        }
    };
#endif

    // ---- pipeline prologue: stage tile 0 ----
    const int nk = K / BK;
    loadA(0);
#if USE_TDM
    if (tid < 32) tdmB(0, 0);
#else
    loadB(0, 0);
#endif
    storeA(0);
#if USE_TDM
    if (tid < 32) {
#if __has_builtin(__builtin_amdgcn_s_wait_tensorcnt)
        __builtin_amdgcn_s_wait_tensorcnt(0);
#endif
    }
#endif

    for (int i = 0; i < nk; ++i) {
        const int buf = i & 1;
        __syncthreads();                      // publish tile i (A stores + TDM)

        const bool hasNext = (i + 1) < nk;    // uniform
        if (hasNext) {
            loadA((i + 1) * BK);              // batched global loads (no wait yet)
#if USE_TDM
            if (tid < 32) tdmB((i + 1) * BK, buf ^ 1);
#else
            loadB((i + 1) * BK, buf ^ 1);
#endif
        }

        // ---- compute tile i (ISA 7.12.2 wave32 fragment layouts) ----
        // A 16x32 f16: lane (hi,r): row r, K = hi*8 + {0..7} and +16
        const _Float16* ap = &Ash[buf][wv * 16 + r][hi * 8];
        v8h a0 = *(const v8h*)(ap);
        v8h a1 = *(const v8h*)(ap + 16);
        v16h af;
        #pragma unroll
        for (int j = 0; j < 8; ++j) { af[j] = a0[j]; af[8 + j] = a1[j]; }

        #pragma unroll
        for (int nt = 0; nt < 4; ++nt) {
            // B 32x16 f16: lane (hi,r): col r, K = hi*8 + {0..7} and +16
            const _Float16* bp = &Bsh[buf][nt * 16 + r][hi * 8];
            v8h b0 = *(const v8h*)(bp);
            v8h b1 = *(const v8h*)(bp + 16);
            v16h bf;
            #pragma unroll
            for (int j = 0; j < 8; ++j) { bf[j] = b0[j]; bf[8 + j] = b1[j]; }
            acc[nt] = __builtin_amdgcn_wmma_f32_16x16x32_f16(
                false, af, false, bf, (short)0, acc[nt], false, false);
        }

        if (hasNext) {
            storeA(buf ^ 1);                  // single loadcnt wait, cvt, ds store
#if USE_TDM
            if (tid < 32) {
#if __has_builtin(__builtin_amdgcn_s_wait_tensorcnt)
                __builtin_amdgcn_s_wait_tensorcnt(0);
#endif
            }
#endif
        }
    }

    // ---- epilogue: C/D layout = lane col r, rows (hi*8 + i) ----
    #pragma unroll
    for (int nt = 0; nt < 4; ++nt) {
        int col = n0 + nt * 16 + r;
        if (col >= N) continue;
        float bv = bias ? bias[col] : 0.0f;
        #pragma unroll
        for (int i = 0; i < 8; ++i) {
            int row = m0 + wv * 16 + hi * 8 + i;
            float* cp = C + (size_t)row * ldc + col;
            float v = acc[nt][i] + bv;
            if (ACC) v += *cp;
            if (ACT == 1)      v = fmaxf(v, 0.0f);
            else if (ACT == 2) v = tanhf(v);
            *cp = v;
        }
    }
}

// ------------------------------------------------------ elementwise kernels
__global__ void k_zero(float* p, int n) {
    int i = blockIdx.x * 256 + threadIdx.x;
    if (i < n) p[i] = 0.0f;
}

__global__ void k_transpose_f16(const float* __restrict__ W, _Float16* __restrict__ Wt,
                                int K, int N) {  // W[K][N] -> Wt[N][K]
    int idx = blockIdx.x * 256 + threadIdx.x;
    if (idx >= K * N) return;
    int n = idx / K, k = idx - n * K;
    Wt[(size_t)n * K + k] = (_Float16)W[(size_t)k * N + n];
}

__global__ void k_concat3(const float* a, const float* b, const float* c,
                          float* out, int n) {
    int i = blockIdx.x * 256 + threadIdx.x;
    if (i >= 3 * n) return;
    out[i] = (i < n) ? a[i] : ((i < 2 * n) ? b[i - n] : c[i - 2 * n]);
}

__global__ void k_gate_rh(const float* __restrict__ G, const float* __restrict__ h,
                          float* __restrict__ rh, int rows) {
    int i = blockIdx.x * 256 + threadIdx.x;
    if (i >= rows * cH) return;
    int n = i >> 9, j = i & (cH - 1);
    rh[i] = sigmoidf_(G[(size_t)n * cLDG + j]) * h[i];
}

__global__ void k_euler_step(const float* __restrict__ G, const float* __restrict__ hc,
                             float* __restrict__ hn, const float* __restrict__ times,
                             int t, int rows) {
    int i = blockIdx.x * 256 + threadIdx.x;
    if (i >= rows * cH) return;
    float dt = (t == 0) ? times[0] : (times[t] - times[t - 1]);
    int n = i >> 9, j = i & (cH - 1);
    float z = sigmoidf_(G[(size_t)n * cLDG + cH + j]);
    float u = tanhf(G[(size_t)n * cLDG + 2 * cH + j]);
    float h = hc[i];
    hn[i] = h + dt * (1.0f - z) * (u - h);
}

__global__ void k_gather(const float* __restrict__ h_all, const int* __restrict__ obs_idx,
                         const int* __restrict__ tix, float* __restrict__ h_pre,
                         int NOBS) {
    int i = blockIdx.x * 256 + threadIdx.x;
    if (i >= NOBS * cH) return;
    int n = i >> 9, j = i & (cH - 1);
    int t = tix[n] + 1;
    h_pre[i] = h_all[((size_t)t * cB + obs_idx[n]) * cH + j];
}

__global__ void k_recon(const float* __restrict__ Ppre, const float* __restrict__ X,
                        const float* __restrict__ Msk, float* __restrict__ gi4,
                        float* accum, int NOBS) {
    int i = blockIdx.x * 256 + threadIdx.x;
    float term = 0.0f;
    if (i < NOBS * cD) {
        int n = i >> 6, d = i & (cD - 1);
        float mean = Ppre[(size_t)n * 2 * cD + d];
        float lv   = Ppre[(size_t)n * 2 * cD + cD + d];
        float x = X[i], m = Msk[i];
        float sigma = __expf(0.5f * lv);
        float err = (x - mean) / sigma;
        term = (0.5f * (err * err + lv) + 0.9189385332046727f) * m;
        float* g = gi4 + (size_t)i * 4;
        g[0] = x; g[1] = mean; g[2] = lv; g[3] = err;
    }
    blockReduceAtomicAdd(term, accum);
}

__global__ void k_prep(const float* __restrict__ gi4, const float* __restrict__ w_prep,
                       const float* __restrict__ b_prep, const float* __restrict__ Msk,
                       float* __restrict__ gi, int NOBS) {
    int i = blockIdx.x * 256 + threadIdx.x;
    if (i >= NOBS * cD * cP) return;
    int p = i & (cP - 1); int t = i >> 6; int d = t & (cD - 1); int n = t >> 6;
    const float* g = gi4 + ((size_t)n * cD + d) * 4;
    const float* w = w_prep + (size_t)d * 4 * cP + p;
    float s = b_prep[d * cP + p]
            + g[0] * w[0] + g[1] * w[cP] + g[2] * w[2 * cP] + g[3] * w[3 * cP];
    s = fmaxf(s, 0.0f) * Msk[(size_t)n * cD + d];
    gi[(size_t)n * (cD * cP) + d * cP + p] = s;
}

__global__ void k_grucell(const float* __restrict__ Gih, const float* __restrict__ Ghh,
                          const float* __restrict__ h_pre, float* __restrict__ h_post,
                          int NOBS) {
    int i = blockIdx.x * 256 + threadIdx.x;
    if (i >= NOBS * cH) return;
    int n = i >> 9, j = i & (cH - 1);
    const float* gi_ = Gih + (size_t)n * cLDG;
    const float* gh_ = Ghh + (size_t)n * cLDG;
    float r  = sigmoidf_(gi_[j] + gh_[j]);
    float z  = sigmoidf_(gi_[cH + j] + gh_[cH + j]);
    float nn = tanhf(gi_[2 * cH + j] + r * gh_[2 * cH + j]);
    float hp = h_pre[i];
    h_post[i] = (1.0f - z) * nn + z * hp;
}

__global__ void k_kl(const float* __restrict__ P2, const float* __restrict__ X,
                     const float* __restrict__ Msk, float* accum, int NOBS) {
    int i = blockIdx.x * 256 + threadIdx.x;
    float term = 0.0f;
    if (i < NOBS * cD) {
        int n = i >> 6, d = i & (cD - 1);
        float m2  = P2[(size_t)n * 2 * cD + d];
        float lv2 = P2[(size_t)n * 2 * cD + cD + d];
        float x = X[i];
        float var2 = __expf(lv2);
        float kl = logf(0.01f) - 0.5f * lv2
                 + (var2 + (m2 - x) * (m2 - x)) / (2.0f * 0.0001f) - 0.5f;
        term = kl * Msk[i];
    }
    blockReduceAtomicAdd(term, accum);
}

__global__ void k_bce(const float* __restrict__ logits, const float* __restrict__ labels,
                      const int* __restrict__ idx, float* accum, int NR) {
    int i = blockIdx.x * 256 + threadIdx.x;
    float term = 0.0f;
    if (i < NR * cCO) {
        int n = i >> 5, c = i & (cCO - 1);
        float l = logits[i];
        int row = idx ? idx[n] : n;
        float y = labels[(size_t)row * cCO + c];
        term = fmaxf(l, 0.0f) - l * y + log1pf(__expf(-fabsf(l)));
    }
    blockReduceAtomicAdd(term, accum);
}

__global__ void k_scatter_jump(const float* __restrict__ h_post, const float* __restrict__ h_pre,
                               const int* __restrict__ obs_idx, float* __restrict__ h_jump,
                               int NOBS) {
    int i = blockIdx.x * 256 + threadIdx.x;
    if (i >= NOBS * cH) return;
    int n = i >> 9, j = i & (cH - 1);
    atomicAdd(&h_jump[(size_t)obs_idx[n] * cH + j], h_post[i] - h_pre[i]);
}

__global__ void k_hfinal(const float* __restrict__ h_last, const float* __restrict__ h_jump,
                         float* __restrict__ out, int n) {
    int i = blockIdx.x * 256 + threadIdx.x;
    if (i < n) out[i] = h_last[i] + 2.0f * h_jump[i];
}

__global__ void k_finalize(const float* accum, float* out_total, float* out_class) {
    if (threadIdx.x == 0 && blockIdx.x == 0) {
        float cls = accum[2];
        *out_total = accum[0] + accum[1] + cls;
        *out_class = cls;
    }
}

// ----------------------------------------------------------- host helpers --
static inline void launch_gemm(hipStream_t s, const float* A, const _Float16* Bt,
                               const float* bias, float* C,
                               int M, int N, int K, int ldc, int act, bool acc) {
    dim3 grid((N + BN - 1) / BN, (M + BM - 1) / BM), blk(256);
    if (acc) {
        wmma_gemm_kernel<0, true><<<grid, blk, 0, s>>>(A, Bt, bias, C, M, N, K, ldc);
    } else {
        switch (act) {
            case 1:  wmma_gemm_kernel<1, false><<<grid, blk, 0, s>>>(A, Bt, bias, C, M, N, K, ldc); break;
            case 2:  wmma_gemm_kernel<2, false><<<grid, blk, 0, s>>>(A, Bt, bias, C, M, N, K, ldc); break;
            default: wmma_gemm_kernel<0, false><<<grid, blk, 0, s>>>(A, Bt, bias, C, M, N, K, ldc); break;
        }
    }
}

static inline void launch_transpose(hipStream_t s, const float* W, _Float16* Wt, int K, int N) {
    int n = K * N;
    k_transpose_f16<<<(n + 255) / 256, 256, 0, s>>>(W, Wt, K, N);
}

// -------------------------------------------------------------- entry ------
extern "C" void kernel_launch(void* const* d_in, const int* in_sizes, int n_in,
                              void* d_out, int out_size, void* d_ws, size_t ws_size,
                              hipStream_t stream) {
    (void)in_sizes; (void)n_in; (void)out_size; (void)ws_size;

    // dims
    const int B = 2048, H = 512, D = 64, PH = 512, PREP = 64;
    const int COV = 128, COVH = 512, CH = 512, CO = 32;
    const int NT = 128, NOBS = 16384, DI = 2 * D, GIN = PREP * D;

    // inputs (setup_inputs dict order, params in insertion order)
    const float* times    = (const float*)d_in[0];
    const float* X        = (const float*)d_in[1];
    const float* Msk      = (const float*)d_in[2];
    const float* cov      = (const float*)d_in[3];
    const float* labels   = (const float*)d_in[4];
    const int*   obs_idx  = (const int*)d_in[5];
    const int*   tix      = (const int*)d_in[6];
    const float* cov_W1   = (const float*)d_in[7];
    const float* cov_b1   = (const float*)d_in[8];
    const float* cov_W2   = (const float*)d_in[9];
    const float* cov_b2   = (const float*)d_in[10];
    const float* p_W1     = (const float*)d_in[11];
    const float* p_b1     = (const float*)d_in[12];
    const float* p_W2     = (const float*)d_in[13];
    const float* p_b2     = (const float*)d_in[14];
    const float* xr_W     = (const float*)d_in[15];
    const float* xr_b     = (const float*)d_in[16];
    const float* xz_W     = (const float*)d_in[17];
    const float* xz_b     = (const float*)d_in[18];
    const float* xh_W     = (const float*)d_in[19];
    const float* xh_b     = (const float*)d_in[20];
    const float* hr_W     = (const float*)d_in[21];
    const float* hz_W     = (const float*)d_in[22];
    const float* hh_W     = (const float*)d_in[23];
    const float* w_prep   = (const float*)d_in[24];
    const float* b_prep   = (const float*)d_in[25];
    const float* gru_Wih  = (const float*)d_in[26];
    const float* gru_bih  = (const float*)d_in[27];
    const float* gru_Whh  = (const float*)d_in[28];
    const float* gru_bhh  = (const float*)d_in[29];
    const float* cls_W1   = (const float*)d_in[30];
    const float* cls_b1   = (const float*)d_in[31];
    const float* cls_W2   = (const float*)d_in[32];
    const float* cls_b2   = (const float*)d_in[33];

    // workspace carve-out
    size_t off = 0;
    auto alloc = [&](size_t bytes) {
        void* p = (char*)d_ws + off;
        off += (bytes + 255) & ~(size_t)255;
        return p;
    };
    float*    accum    = (float*)alloc(8 * sizeof(float));
    _Float16* pW1t     = (_Float16*)alloc((size_t)PH * H * 2);
    _Float16* pW2t     = (_Float16*)alloc((size_t)2 * D * PH * 2);
    _Float16* covW1t   = (_Float16*)alloc((size_t)COVH * COV * 2);
    _Float16* covW2t   = (_Float16*)alloc((size_t)H * COVH * 2);
    _Float16* Wxcat    = (_Float16*)alloc((size_t)3 * H * DI * 2);   // [1536][128]
    _Float16* Whrz     = (_Float16*)alloc((size_t)2 * H * H * 2);    // [1024][512]
    _Float16* hhWt     = (_Float16*)alloc((size_t)H * H * 2);
    _Float16* Wiht     = (_Float16*)alloc((size_t)3 * H * GIN * 2);
    _Float16* Whht     = (_Float16*)alloc((size_t)3 * H * H * 2);
    _Float16* clsW1t   = (_Float16*)alloc((size_t)CH * H * 2);
    _Float16* clsW2t   = (_Float16*)alloc((size_t)CO * CH * 2);
    float*    b_gates  = (float*)alloc((size_t)3 * H * sizeof(float));
    float*    h_all    = (float*)alloc((size_t)(NT + 1) * B * H * sizeof(float));
    float*    tmp512   = (float*)alloc((size_t)B * H * sizeof(float));
    float*    xbuf     = (float*)alloc((size_t)B * DI * sizeof(float));
    float*    G        = (float*)alloc((size_t)B * 3 * H * sizeof(float));
    float*    rh       = (float*)alloc((size_t)B * H * sizeof(float));
    float*    h_pre    = (float*)alloc((size_t)NOBS * H * sizeof(float));
    float*    A1       = (float*)alloc((size_t)NOBS * H * sizeof(float));
    float*    Ppre     = (float*)alloc((size_t)NOBS * 2 * D * sizeof(float));
    float*    gi4      = (float*)alloc((size_t)NOBS * D * 4 * sizeof(float));
    float*    gi       = (float*)alloc((size_t)NOBS * GIN * sizeof(float));
    float*    Gih      = (float*)alloc((size_t)NOBS * 3 * H * sizeof(float));
    float*    Ghh      = (float*)alloc((size_t)NOBS * 3 * H * sizeof(float));
    float*    h_post   = (float*)alloc((size_t)NOBS * H * sizeof(float));
    float*    P2       = (float*)alloc((size_t)NOBS * 2 * D * sizeof(float));
    float*    pred_obs = (float*)alloc((size_t)NOBS * CO * sizeof(float));
    float*    h_jump   = (float*)alloc((size_t)B * H * sizeof(float));

    hipStream_t s = stream;
    float* out = (float*)d_out;

    // ---- setup: zero accumulators, convert+transpose weights to f16 ----
    k_zero<<<1, 256, 0, s>>>(accum, 8);
    launch_transpose(s, p_W1, pW1t, H, PH);
    launch_transpose(s, p_W2, pW2t, PH, 2 * D);
    launch_transpose(s, cov_W1, covW1t, COV, COVH);
    launch_transpose(s, cov_W2, covW2t, COVH, H);
    launch_transpose(s, xr_W, Wxcat,               DI, H);   // rows 0..511
    launch_transpose(s, xz_W, Wxcat + (size_t)H * DI,     DI, H);
    launch_transpose(s, xh_W, Wxcat + (size_t)2 * H * DI, DI, H);
    launch_transpose(s, hr_W, Whrz,                H, H);
    launch_transpose(s, hz_W, Whrz + (size_t)H * H, H, H);
    launch_transpose(s, hh_W, hhWt, H, H);
    launch_transpose(s, gru_Wih, Wiht, GIN, 3 * H);
    launch_transpose(s, gru_Whh, Whht, H, 3 * H);
    launch_transpose(s, cls_W1, clsW1t, H, CH);
    launch_transpose(s, cls_W2, clsW2t, CH, CO);
    k_concat3<<<(3 * H + 255) / 256, 256, 0, s>>>(xr_b, xz_b, xh_b, b_gates, H);

    // ---- cov_model -> h0 ----
    launch_gemm(s, cov, covW1t, cov_b1, tmp512, B, COVH, COV, COVH, 1, false);
    launch_gemm(s, tmp512, covW2t, cov_b2, h_all, B, H, COVH, H, 2, false);

    // ---- Euler scan: 128 steps of fused GRU-ODE ----
    const int EW = (B * H + 255) / 256;
    for (int t = 0; t < NT; ++t) {
        const float* hc = h_all + (size_t)t * B * H;
        float*       hn = h_all + (size_t)(t + 1) * B * H;
        launch_gemm(s, hc, pW1t, p_b1, tmp512, B, PH, H, PH, 1, false);          // relu(h@p_W1+b)
        launch_gemm(s, tmp512, pW2t, p_b2, xbuf, B, 2 * D, PH, 2 * D, 0, false); // x
        launch_gemm(s, xbuf, Wxcat, b_gates, G, B, 3 * H, DI, 3 * H, 0, false);  // x-gates
        launch_gemm(s, hc, Whrz, nullptr, G, B, 2 * H, H, 3 * H, 0, true);       // +h@[hr|hz]
        k_gate_rh<<<EW, 256, 0, s>>>(G, hc, rh, B);                              // rh = sig(r)*h
        launch_gemm(s, rh, hhWt, nullptr, G + 2 * H, B, H, H, 3 * H, 0, true);   // +rh@hh
        k_euler_step<<<EW, 256, 0, s>>>(G, hc, hn, times, t, B);
    }

    // ---- observation phase ----
    const int OW = (NOBS * H + 255) / 256;
    k_gather<<<OW, 256, 0, s>>>(h_all, obs_idx, tix, h_pre, NOBS);
    launch_gemm(s, h_pre, pW1t, p_b1, A1, NOBS, PH, H, PH, 1, false);
    launch_gemm(s, A1, pW2t, p_b2, Ppre, NOBS, 2 * D, PH, 2 * D, 0, false);
    k_recon<<<(NOBS * D + 255) / 256, 256, 0, s>>>(Ppre, X, Msk, gi4, accum + 0, NOBS);
    k_prep<<<(NOBS * D * PREP + 255) / 256, 256, 0, s>>>(gi4, w_prep, b_prep, Msk, gi, NOBS);
    launch_gemm(s, gi, Wiht, gru_bih, Gih, NOBS, 3 * H, GIN, 3 * H, 0, false);   // big GEMM
    launch_gemm(s, h_pre, Whht, gru_bhh, Ghh, NOBS, 3 * H, H, 3 * H, 0, false);
    k_grucell<<<OW, 256, 0, s>>>(Gih, Ghh, h_pre, h_post, NOBS);
    launch_gemm(s, h_post, pW1t, p_b1, A1, NOBS, PH, H, PH, 1, false);
    launch_gemm(s, A1, pW2t, p_b2, P2, NOBS, 2 * D, PH, 2 * D, 0, false);
    k_kl<<<(NOBS * D + 255) / 256, 256, 0, s>>>(P2, X, Msk, accum + 1, NOBS);
    launch_gemm(s, h_post, clsW1t, cls_b1, A1, NOBS, CH, H, CH, 1, false);
    launch_gemm(s, A1, clsW2t, cls_b2, pred_obs, NOBS, CO, CH, CO, 0, false);
    k_bce<<<(NOBS * CO + 255) / 256, 256, 0, s>>>(pred_obs, labels, obs_idx, accum + 2, NOBS);

    // ---- jump scatter + final head ----
    k_zero<<<EW, 256, 0, s>>>(h_jump, B * H);
    k_scatter_jump<<<OW, 256, 0, s>>>(h_post, h_pre, obs_idx, h_jump, NOBS);
    k_hfinal<<<EW, 256, 0, s>>>(h_all + (size_t)NT * B * H, h_jump, out, B * H);
    launch_gemm(s, out, clsW1t, cls_b1, tmp512, B, CH, H, CH, 1, false);
    launch_gemm(s, tmp512, clsW2t, cls_b2, out + (size_t)B * H + 1, B, CO, CH, CO, 0, false);
    k_bce<<<(B * CO + 255) / 256, 256, 0, s>>>(out + (size_t)B * H + 1, labels, nullptr, accum + 2, B);
    k_finalize<<<1, 32, 0, s>>>(accum, out + (size_t)B * H,
                                out + (size_t)B * H + 1 + (size_t)B * CO);
}
